// Hawk_66881230734025
// MI455X (gfx1250) — matrix operand order repaired and checked by reading, not compiled, not run on hardware
//
#include <hip/hip_runtime.h>
#include <hip/hip_bf16.h>
#include <math.h>

typedef __bf16 bf16_t;
typedef __attribute__((ext_vector_type(16))) __bf16 v16bf;
typedef __attribute__((ext_vector_type(8)))  float  v8f;

#define DIM   1024
#define HID   1536
#define KCONV 4
#define BATCH 2
#define TLEN  4096
#define BT    (BATCH * TLEN)       // 8192
#define GU_LD (2 * HID)            // 3072

#define TC     256                 // scan chunk length
#define NCHUNK (TLEN / TC)         // 16
#define NCHAN  (BATCH * HID)       // 3072

// ---------------------------------------------------------------------------
// f32 -> (bf16 hi, bf16 lo) split:  x ~= hi + lo, lo = bf16(x - float(hi))
// ---------------------------------------------------------------------------
__device__ __forceinline__ void split_bf16(float x, bf16_t& hi, bf16_t& lo) {
  bf16_t h = (bf16_t)x;
  hi = h;
  lo = (bf16_t)(x - (float)h);
}

__global__ void cvt_split_f32_bf16(const float* __restrict__ src,
                                   bf16_t* __restrict__ hi,
                                   bf16_t* __restrict__ lo, int n) {
  int i = blockIdx.x * blockDim.x + threadIdx.x;
  if (i < n) {
    bf16_t h, l;
    split_bf16(src[i], h, l);
    hi[i] = h;
    lo[i] = l;
  }
}

// ---------------------------------------------------------------------------
// Split-precision GEMM: C[M,N] = A[M,K](f32, row stride lda) * W[N,K]^T
// A split in-register to bf16 hi/lo; W pre-split.  3-term bf16x3 products:
//   acc += Ahi*Whi + Alo*Whi + Ahi*Wlo   (f32 accumulate)  ~ fp32 accuracy.
// One wave computes a 64x64 tile; 48 v_wmma_f32_16x16x32_bf16 per K-step 32.
// A/B fragments use contiguous K, k = lh*16 + e, in both (wave32).
// ---------------------------------------------------------------------------
__global__ __launch_bounds__(32)
void gemm_bf16x3_wmma(const float* __restrict__ A, int lda,
                      const bf16_t* __restrict__ Whi,
                      const bf16_t* __restrict__ Wlo,
                      float* __restrict__ C, int ldc,
                      int M, int N, int K) {
  const int lane = threadIdx.x;
  const int m0 = blockIdx.y * 64;
  const int n0 = blockIdx.x * 64;
  const int lr = lane & 15;   // row (A) / col (B) within 16
  const int lh = lane >> 4;   // K-half selector

  v8f acc[4][4] = {};

  const float* Abase = A + (size_t)(m0 + lr) * lda;                // + mi*16*lda
  const size_t Woff  = (size_t)(n0 + lr) * K + lh * 16;            // + ni*16*K

  for (int k0 = 0; k0 < K; k0 += 32) {
    // ---- A fragments: 16x32 bf16, lane = row lr, K contiguous at lh*16
    v16bf ah[4], al[4];
#pragma unroll
    for (int mi = 0; mi < 4; ++mi) {
      const float* p0 = Abase + (size_t)(mi * 16) * lda + k0 + lh * 16;
      float4 f0 = *(const float4*)(p0);
      float4 f1 = *(const float4*)(p0 + 4);
      float4 f2 = *(const float4*)(p0 + 8);
      float4 f3 = *(const float4*)(p0 + 12);
      float fe[16] = {f0.x, f0.y, f0.z, f0.w, f1.x, f1.y, f1.z, f1.w,
                      f2.x, f2.y, f2.z, f2.w, f3.x, f3.y, f3.z, f3.w};
      v16bf h, l;
#pragma unroll
      for (int e = 0; e < 16; ++e) {
        bf16_t eh, el;
        split_bf16(fe[e], eh, el);
        h[e] = eh;
        l[e] = el;
      }
      ah[mi] = h;
      al[mi] = l;
    }
    // ---- B subtiles one at a time to bound VGPR pressure
#pragma unroll
    for (int ni = 0; ni < 4; ++ni) {
      const size_t bo = Woff + (size_t)(ni * 16) * K + k0;
      v16bf bh = *(const v16bf*)(Whi + bo);
      v16bf bl = *(const v16bf*)(Wlo + bo);
#pragma unroll
      for (int mi = 0; mi < 4; ++mi) {
        acc[mi][ni] = __builtin_amdgcn_wmma_f32_16x16x32_bf16(
            false, ah[mi], false, bh, (short)0, acc[mi][ni], false, false);
        acc[mi][ni] = __builtin_amdgcn_wmma_f32_16x16x32_bf16(
            false, al[mi], false, bh, (short)0, acc[mi][ni], false, false);
        acc[mi][ni] = __builtin_amdgcn_wmma_f32_16x16x32_bf16(
            false, ah[mi], false, bl, (short)0, acc[mi][ni], false, false);
      }
    }
  }

  // ---- epilogue: v8f reg r -> row lh*8 + r, col lr (ISA C/D layout)
#pragma unroll
  for (int mi = 0; mi < 4; ++mi)
#pragma unroll
    for (int ni = 0; ni < 4; ++ni)
#pragma unroll
      for (int r = 0; r < 8; ++r) {
        int row = m0 + mi * 16 + lh * 8 + r;
        int col = n0 + ni * 16 + lr;
        C[(size_t)row * ldc + col] = acc[mi][ni][r];
      }
}

// ---------------------------------------------------------------------------
// Causal depthwise conv1d (K=4): u[bt,h] = sum_k u_pre[t-3+k,h]*cw[h,k] + cb[h]
// u_pre lives in gu columns [HID, 2*HID)
// ---------------------------------------------------------------------------
__global__ void conv_kernel(const float* __restrict__ gu,
                            const float* __restrict__ cw,
                            const float* __restrict__ cb,
                            float* __restrict__ u) {
  size_t i = (size_t)blockIdx.x * blockDim.x + threadIdx.x;
  if (i >= (size_t)BT * HID) return;
  int h = (int)(i % HID);
  size_t bt = i / HID;
  int t = (int)(bt % TLEN);
  const float* base = gu + bt * (size_t)GU_LD + HID + h;
  float acc = cb[h];
#pragma unroll
  for (int k = 0; k < KCONV; ++k) {
    int dt = t - (KCONV - 1) + k;
    if (dt >= 0) acc += base[(long)(k - (KCONV - 1)) * GU_LD] * cw[h * KCONV + k];
  }
  u[i] = acc;
}

// ---------------------------------------------------------------------------
// RG-LRU elementwise: alpha -> rgb (in place), xbeta -> igb (in place)
// ---------------------------------------------------------------------------
__global__ void rglru_kernel(float* __restrict__ igb, float* __restrict__ rgb,
                             const float* __restrict__ u,
                             const float* __restrict__ fl,
                             const float* __restrict__ b_in,
                             const float* __restrict__ b_gate) {
  size_t i = (size_t)blockIdx.x * blockDim.x + threadIdx.x;
  if (i >= (size_t)BT * HID) return;
  int g = (int)(i % HID);
  float ig = igb[i] + b_in[g];
  float rg = rgb[i] + b_gate[g];
  float uu = u[i];
  float sp    = log1pf(expf(fl[g]));              // softplus(forget_lambda)
  float srg   = 1.0f / (1.0f + expf(-rg));
  float alpha = expf(-8.0f * sp * srg);
  float beta  = sqrtf(1.0f - alpha * alpha + 1e-6f);
  float sig   = 1.0f / (1.0f + expf(-ig));
  rgb[i] = alpha;
  igb[i] = beta * sig * uu;
}

// ---------------------------------------------------------------------------
// Scan phase 1: per (channel, chunk) inclusive scan.
// alpha buffer <- running product P_t ; x buffer <- local scan L_t.
// ---------------------------------------------------------------------------
__global__ void scan_phase1(float* __restrict__ alpha, float* __restrict__ x,
                            float* __restrict__ carryA, float* __restrict__ carryH) {
  int tid = blockIdx.x * blockDim.x + threadIdx.x;
  if (tid >= NCHAN * NCHUNK) return;
  int c  = tid % NCHAN;
  int ck = tid / NCHAN;
  int b = c / HID, h = c % HID;
  size_t base = ((size_t)(b * TLEN + ck * TC)) * HID + h;
  float P = 1.0f, L = 0.0f;
  for (int i = 0; i < TC; ++i) {
    float a = alpha[base];
    float v = x[base];
    L = a * L + v;
    P *= a;
    alpha[base] = P;
    x[base] = L;
    base += HID;
  }
  carryA[c * NCHUNK + ck] = P;
  carryH[c * NCHUNK + ck] = L;
}

// Scan phase 2: fold preceding chunk carries, apply h_t = L_t + P_t * h_in.
__global__ void scan_phase2(const float* __restrict__ alphaP, float* __restrict__ x,
                            const float* __restrict__ carryA,
                            const float* __restrict__ carryH) {
  int tid = blockIdx.x * blockDim.x + threadIdx.x;
  if (tid >= NCHAN * NCHUNK) return;
  int c  = tid % NCHAN;
  int ck = tid / NCHAN;
  if (ck == 0) return;
  float hin = 0.0f;
  for (int j = 0; j < ck; ++j)
    hin = carryA[c * NCHUNK + j] * hin + carryH[c * NCHUNK + j];
  int b = c / HID, h = c % HID;
  size_t base = ((size_t)(b * TLEN + ck * TC)) * HID + h;
  for (int i = 0; i < TC; ++i) {
    x[base] += alphaP[base] * hin;
    base += HID;
  }
}

// ---------------------------------------------------------------------------
// gelu(gate) * h, written in place into the gate columns of gu.
// ---------------------------------------------------------------------------
__global__ void gelugate_kernel(float* __restrict__ gu, const float* __restrict__ h) {
  size_t i = (size_t)blockIdx.x * blockDim.x + threadIdx.x;
  if (i >= (size_t)BT * HID) return;
  size_t bt = i / HID;
  int g = (int)(i % HID);
  float* gp = gu + bt * (size_t)GU_LD + g;
  float gv = *gp;
  float ge = 0.5f * gv * (1.0f + erff(gv * 0.70710678118654752f));
  *gp = ge * h[i];
}

// ---------------------------------------------------------------------------
// launch
// ---------------------------------------------------------------------------
extern "C" void kernel_launch(void* const* d_in, const int* in_sizes, int n_in,
                              void* d_out, int out_size, void* d_ws, size_t ws_size,
                              hipStream_t stream) {
  (void)in_sizes; (void)n_in; (void)out_size; (void)ws_size;
  const float* x       = (const float*)d_in[0];
  const float* W_proj  = (const float*)d_in[1];
  const float* conv_w  = (const float*)d_in[2];
  const float* conv_b  = (const float*)d_in[3];
  const float* W_in    = (const float*)d_in[4];
  const float* b_in    = (const float*)d_in[5];
  const float* W_gate  = (const float*)d_in[6];
  const float* b_gate  = (const float*)d_in[7];
  const float* fl      = (const float*)d_in[8];
  const float* W_out   = (const float*)d_in[9];
  float* out = (float*)d_out;

  // workspace layout
  char* ws = (char*)d_ws;
  const size_t SZ_GU = (size_t)BT * GU_LD * sizeof(float);   // 100.7 MB
  const size_t SZ_H  = (size_t)BT * HID * sizeof(float);     // 50.3 MB
  const size_t SZ_WP = (size_t)GU_LD * DIM * sizeof(bf16_t); // 6.3 MB
  const size_t SZ_WH = (size_t)HID * HID * sizeof(bf16_t);   // 4.7 MB
  const size_t SZ_WO = (size_t)DIM * HID * sizeof(bf16_t);   // 3.1 MB

  float*  gu = (float*)(ws);
  float*  u  = (float*)(ws + SZ_GU);
  float*  ig = (float*)(ws + SZ_GU + SZ_H);
  float*  rg = (float*)(ws + SZ_GU + 2 * SZ_H);
  char*   p  = ws + SZ_GU + 3 * SZ_H;
  bf16_t* wp_hi  = (bf16_t*)p;  p += SZ_WP;
  bf16_t* wp_lo  = (bf16_t*)p;  p += SZ_WP;
  bf16_t* win_hi = (bf16_t*)p;  p += SZ_WH;
  bf16_t* win_lo = (bf16_t*)p;  p += SZ_WH;
  bf16_t* wg_hi  = (bf16_t*)p;  p += SZ_WH;
  bf16_t* wg_lo  = (bf16_t*)p;  p += SZ_WH;
  bf16_t* wo_hi  = (bf16_t*)p;  p += SZ_WO;
  bf16_t* wo_lo  = (bf16_t*)p;  p += SZ_WO;
  float* carryA = (float*)p;    p += (size_t)NCHAN * NCHUNK * sizeof(float);
  float* carryH = (float*)p;

  const int EW = 256;
  const size_t NEL = (size_t)BT * HID;
  const int ELB = (int)((NEL + EW - 1) / EW);

  // 1) weights -> split bf16 hi/lo
  cvt_split_f32_bf16<<<(GU_LD * DIM + EW - 1) / EW, EW, 0, stream>>>(
      W_proj, wp_hi, wp_lo, GU_LD * DIM);
  cvt_split_f32_bf16<<<(HID * HID + EW - 1) / EW, EW, 0, stream>>>(
      W_in, win_hi, win_lo, HID * HID);
  cvt_split_f32_bf16<<<(HID * HID + EW - 1) / EW, EW, 0, stream>>>(
      W_gate, wg_hi, wg_lo, HID * HID);
  cvt_split_f32_bf16<<<(DIM * HID + EW - 1) / EW, EW, 0, stream>>>(
      W_out, wo_hi, wo_lo, DIM * HID);

  dim3 wave(32, 1, 1);
  // 2) proj GEMM: gu[BT,3072] = x * W_proj^T
  gemm_bf16x3_wmma<<<dim3(GU_LD / 64, BT / 64), wave, 0, stream>>>(
      x, DIM, wp_hi, wp_lo, gu, GU_LD, BT, GU_LD, DIM);
  // 3) causal depthwise conv -> u
  conv_kernel<<<ELB, EW, 0, stream>>>(gu, conv_w, conv_b, u);
  // 4) gate GEMMs
  gemm_bf16x3_wmma<<<dim3(HID / 64, BT / 64), wave, 0, stream>>>(
      u, HID, win_hi, win_lo, ig, HID, BT, HID, HID);
  gemm_bf16x3_wmma<<<dim3(HID / 64, BT / 64), wave, 0, stream>>>(
      u, HID, wg_hi, wg_lo, rg, HID, BT, HID, HID);
  // 5) RG-LRU elementwise: rg <- alpha, ig <- xbeta
  rglru_kernel<<<ELB, EW, 0, stream>>>(ig, rg, u, fl, b_in, b_gate);
  // 6) chunked scan over T: ig <- h
  scan_phase1<<<(NCHAN * NCHUNK + EW - 1) / EW, EW, 0, stream>>>(rg, ig, carryA, carryH);
  scan_phase2<<<(NCHAN * NCHUNK + EW - 1) / EW, EW, 0, stream>>>(rg, ig, carryA, carryH);
  // 7) gelu(gate) * h in place into gu's gate columns
  gelugate_kernel<<<ELB, EW, 0, stream>>>(gu, ig);
  // 8) output GEMM: out[BT,DIM] = (gelu(gate)*h) * W_out^T   (A strided in gu)
  gemm_bf16x3_wmma<<<dim3(DIM / 64, BT / 64), wave, 0, stream>>>(
      gu, GU_LD, wo_hi, wo_lo, out, DIM, BT, DIM, HID);
}
